// VGAE_10634339025358
// MI455X (gfx1250) — matrix operand order repaired and checked by reading, not compiled, not run on hardware
//
#include <hip/hip_runtime.h>
#include <hip/hip_bf16.h>

#define N_NODES 8192
#define N_EDGES 262144
#define IN_F    256
#define HID     128
#define NT_TILES (N_NODES / 16)

typedef __attribute__((ext_vector_type(16))) _Float16 v16h;
typedef __attribute__((ext_vector_type(8)))  _Float16 v8h;
typedef __attribute__((ext_vector_type(8)))  float    v8f;

// ---------------- degree ----------------
__global__ void __launch_bounds__(256)
k_degree(const int* __restrict__ dst, float* __restrict__ deg) {
    int e = blockIdx.x * blockDim.x + threadIdx.x;
    if (e < N_EDGES) atomicAdd(&deg[dst[e]], 1.0f);
}

// ---------------- scatter-add of features over edges ----------------
template<int F>
__global__ void __launch_bounds__(256)
k_scatter(const int* __restrict__ src, const int* __restrict__ dst,
          const float* __restrict__ h, float* __restrict__ agg) {
    const unsigned groups = F / 4;
    unsigned t = blockIdx.x * blockDim.x + threadIdx.x;
    if (t >= (unsigned)N_EDGES * groups) return;
    unsigned e = t / groups;
    unsigned g = t % groups;
    const float4 v = *(const float4*)(h + (long long)src[e] * F + g * 4);
    float* a = agg + (long long)dst[e] * F + g * 4;
    atomicAdd(a + 0, v.x);
    atomicAdd(a + 1, v.y);
    atomicAdd(a + 2, v.z);
    atomicAdd(a + 3, v.w);
}

// ---------------- SAGE layer: out = [relu](X@Ws + (AGG/deg)@Wn + b) ----------------
template<int K, int NB>
__global__ void __launch_bounds__(HID)
k_sage(const float* __restrict__ X, const float* __restrict__ AGG,
       const float* __restrict__ deg,
       const float* __restrict__ Ws, const float* __restrict__ Wn,
       const float* __restrict__ b, float* __restrict__ out, int do_relu) {
    __shared__ float xs[NB][K];
    __shared__ float ms[NB][K];
    const int n0 = blockIdx.x * NB;
    const int j  = threadIdx.x;            // 0..127
    for (int nb = 0; nb < NB; ++nb) {
        int n = n0 + nb;
        float dinv = 1.0f / fmaxf(deg[n], 1.0f);
        for (int k = j; k < K; k += HID) {
            xs[nb][k] = X[(long long)n * K + k];
            ms[nb][k] = AGG[(long long)n * K + k] * dinv;
        }
    }
    __syncthreads();
    float acc[NB];
    const float bj = b[j];
#pragma unroll
    for (int nb = 0; nb < NB; ++nb) acc[nb] = bj;
    for (int k = 0; k < K; ++k) {
        float ws = Ws[k * HID + j];
        float wn = Wn[k * HID + j];
#pragma unroll
        for (int nb = 0; nb < NB; ++nb)
            acc[nb] += xs[nb][k] * ws + ms[nb][k] * wn;
    }
#pragma unroll
    for (int nb = 0; nb < NB; ++nb) {
        float v = acc[nb];
        if (do_relu) v = fmaxf(v, 0.0f);
        out[(long long)(n0 + nb) * HID + j] = v;
    }
}

// ---------------- plain linear: out = X@W + b ----------------
template<int K, int NB>
__global__ void __launch_bounds__(HID)
k_linear(const float* __restrict__ X, const float* __restrict__ W,
         const float* __restrict__ b, float* __restrict__ out) {
    __shared__ float xs[NB][K];
    const int n0 = blockIdx.x * NB;
    const int j  = threadIdx.x;
    for (int nb = 0; nb < NB; ++nb)
        for (int k = j; k < K; k += HID)
            xs[nb][k] = X[(long long)(n0 + nb) * K + k];
    __syncthreads();
    float acc[NB];
    const float bj = b[j];
#pragma unroll
    for (int nb = 0; nb < NB; ++nb) acc[nb] = bj;
    for (int k = 0; k < K; ++k) {
        float w = W[k * HID + j];
#pragma unroll
        for (int nb = 0; nb < NB; ++nb) acc[nb] += xs[nb][k] * w;
    }
#pragma unroll
    for (int nb = 0; nb < NB; ++nb)
        out[(long long)(n0 + nb) * HID + j] = acc[nb];
}

// ---------------- pack f32 -> f16 for WMMA; also copy h to output ----------------
__global__ void __launch_bounds__(256)
k_pack(const float* __restrict__ h, const float* __restrict__ hw,
       _Float16* __restrict__ h16, _Float16* __restrict__ hw16,
       float* __restrict__ out_h) {
    int i = blockIdx.x * blockDim.x + threadIdx.x;
    if (i < N_NODES * HID) {
        float a = h[i], c = hw[i];
        h16[i]  = (_Float16)a;
        hw16[i] = (_Float16)c;
        out_h[i] = a;
    }
}

__global__ void __launch_bounds__(256)
k_labels(const int* __restrict__ lab, float* __restrict__ out) {
    int i = blockIdx.x * blockDim.x + threadIdx.x;
    if (i < N_NODES) out[i] = (float)lab[i];
}

// ---------------- big GEMM: S = relu(hW @ h^T), track per-row max ----------------
// One block per 16-row M-tile; 8 waves sweep all N-tiles with register
// double-buffering of B fragments (load tile nt+8 while WMMAs consume tile nt).
__global__ void __launch_bounds__(256)
k_score_wmma(const _Float16* __restrict__ A16,   // hW  [N, HID]
             const _Float16* __restrict__ B16,   // h   [N, HID]  (B = h^T)
             float* __restrict__ S,              // [N, N]
             unsigned int* __restrict__ rowmax) {
    const int mtile = blockIdx.x;            // 0..511
    const int wave  = threadIdx.x >> 5;      // 0..7
    const int lane  = threadIdx.x & 31;
    const int l4    = lane & 15;
    const int hi    = lane >> 4;
    const int m     = mtile * 16 + l4;

    // A-fragments (16x32 f16 layout, ISA 7.12.2): per lane, elems 0-7 = K base+hi*8,
    // elems 8-15 = K base+16+hi*8 -> two contiguous 16B loads per K-step.
    v16h afrag[4];
    const _Float16* arow = A16 + (long long)m * HID;
#pragma unroll
    for (int ks = 0; ks < 4; ++ks) {
        v8h lo  = *(const v8h*)(arow + ks * 32 + hi * 8);
        v8h hiv = *(const v8h*)(arow + ks * 32 + 16 + hi * 8);
#pragma unroll
        for (int i = 0; i < 8; ++i) { afrag[ks][i] = lo[i]; afrag[ks][8 + i] = hiv[i]; }
    }

    float vmax[8];
#pragma unroll
    for (int v = 0; v < 8; ++v) vmax[v] = 0.0f;   // relu() >= 0

    // B (32x16 f16): elems 0-7 = h[n][ks*32 + hi*8 + j], elems 8-15 = h[n][ks*32 + 16 + hi*8 + j]
    auto loadB = [&](int nt, v16h* bf) {
        const _Float16* brow = B16 + (long long)(nt * 16 + l4) * HID;
#pragma unroll
        for (int ks = 0; ks < 4; ++ks) {
            v8h p = *(const v8h*)(brow + ks * 32 + hi * 8);
            v8h q = *(const v8h*)(brow + ks * 32 + 16 + hi * 8);
#pragma unroll
            for (int i = 0; i < 8; ++i) { bf[ks][i] = p[i]; bf[ks][8 + i] = q[i]; }
        }
    };

    auto computeStore = [&](int nt, const v16h* bf) {
        v8f c = {};
#pragma unroll
        for (int ks = 0; ks < 4; ++ks)
            c = __builtin_amdgcn_wmma_f32_16x16x32_f16(
                    false, afrag[ks], false, bf[ks], (short)0, c, false, false);
        // D layout: m = 8*hi + v (within tile), n = l4
        float* sbase = S + (long long)(mtile * 16) * N_NODES + nt * 16 + l4;
#pragma unroll
        for (int v = 0; v < 8; ++v) {
            float s = fmaxf(c[v], 0.0f);               // relu
            sbase[(long long)(8 * hi + v) * N_NODES] = s;
            vmax[v] = fmaxf(vmax[v], s);
        }
    };

    v16h b0[4], b1[4];
    loadB(wave, b0);
    // NT_TILES/16 = 32 iterations per wave; nt+8 is always in range.
    for (int nt = wave; nt < NT_TILES; nt += 16) {
        loadB(nt + 8, b1);                 // prefetch while b0 computes
        computeStore(nt, b0);
        if (nt + 16 < NT_TILES) loadB(nt + 16, b0);
        computeStore(nt + 8, b1);
    }

    // reduce max over the 16 lanes of each half-wave, then one atomic per row
#pragma unroll
    for (int v = 0; v < 8; ++v) {
        float x = vmax[v];
        for (int msk = 1; msk < 16; msk <<= 1)
            x = fmaxf(x, __shfl_xor(x, msk, 32));
        if (l4 == 0) {
            int row = mtile * 16 + 8 * hi + v;
            atomicMax(&rowmax[row], __float_as_uint(x));  // valid: all values >= 0
        }
    }
}

// ---------------- row softmax, exp kept in registers (1 read + 1 write) ----------------
__global__ void __launch_bounds__(256)
k_softmax(float* __restrict__ S, const unsigned int* __restrict__ rowmax) {
    __shared__ float red[8];
    const int row = blockIdx.x;
    const int tid = threadIdx.x;
    const float mx = __uint_as_float(rowmax[row]);
    float* rp = S + (long long)row * N_NODES;
    float e[N_NODES / 256];
    float lsum = 0.0f;
#pragma unroll
    for (int i = 0; i < N_NODES / 256; ++i) {
        float ev = __expf(__builtin_nontemporal_load(rp + tid + i * 256) - mx);
        e[i] = ev;
        lsum += ev;
    }
    for (int msk = 1; msk < 32; msk <<= 1) lsum += __shfl_xor(lsum, msk, 32);
    if ((tid & 31) == 0) red[tid >> 5] = lsum;
    __syncthreads();
    float tot = 0.0f;
#pragma unroll
    for (int w = 0; w < 8; ++w) tot += red[w];
    const float inv = 1.0f / tot;
#pragma unroll
    for (int i = 0; i < N_NODES / 256; ++i)
        __builtin_nontemporal_store(e[i] * inv, rp + tid + i * 256);
}

extern "C" void kernel_launch(void* const* d_in, const int* in_sizes, int n_in,
                              void* d_out, int out_size, void* d_ws, size_t ws_size,
                              hipStream_t stream) {
    const float* inputs  = (const float*)d_in[0];
    const int*   src     = (const int*)d_in[1];
    const int*   dst     = (const int*)d_in[2];
    const int*   labels  = (const int*)d_in[3];
    const float* Wself0  = (const float*)d_in[4];
    const float* Wneigh0 = (const float*)d_in[5];
    const float* b0      = (const float*)d_in[6];
    const float* Wself1  = (const float*)d_in[7];
    const float* Wneigh1 = (const float*)d_in[8];
    const float* b1      = (const float*)d_in[9];
    const float* Wdec    = (const float*)d_in[10];
    const float* bdec    = (const float*)d_in[11];

    float* out_adj = (float*)d_out;                                    // [N, N]
    float* out_h   = out_adj + (long long)N_NODES * N_NODES;           // [N, HID]
    float* out_lab = out_h + (long long)N_NODES * HID;                 // [N]

    // workspace carve-out (256B aligned slices)
    char* ws = (char*)d_ws;
    size_t off = 0;
    auto carve = [&](size_t bytes) -> void* {
        void* p = ws + off;
        off += (bytes + 255) & ~(size_t)255;
        return p;
    };
    float*        deg    = (float*)carve((size_t)N_NODES * 4);
    float*        agg0   = (float*)carve((size_t)N_NODES * IN_F * 4);
    float*        h0     = (float*)carve((size_t)N_NODES * HID * 4);
    float*        agg1   = (float*)carve((size_t)N_NODES * HID * 4);
    float*        h1     = (float*)carve((size_t)N_NODES * HID * 4);
    float*        hw     = (float*)carve((size_t)N_NODES * HID * 4);
    _Float16*     h16    = (_Float16*)carve((size_t)N_NODES * HID * 2);
    _Float16*     hw16   = (_Float16*)carve((size_t)N_NODES * HID * 2);
    unsigned int* rowmax = (unsigned int*)carve((size_t)N_NODES * 4);

    hipMemsetAsync(deg,    0, (size_t)N_NODES * 4, stream);
    hipMemsetAsync(agg0,   0, (size_t)N_NODES * IN_F * 4, stream);
    hipMemsetAsync(agg1,   0, (size_t)N_NODES * HID * 4, stream);
    hipMemsetAsync(rowmax, 0, (size_t)N_NODES * 4, stream);

    // layer 0
    k_degree<<<(N_EDGES + 255) / 256, 256, 0, stream>>>(dst, deg);
    k_scatter<IN_F><<<(N_EDGES * (IN_F / 4) + 255) / 256, 256, 0, stream>>>(src, dst, inputs, agg0);
    k_sage<IN_F, 4><<<N_NODES / 4, HID, 0, stream>>>(inputs, agg0, deg, Wself0, Wneigh0, b0, h0, 1);
    // layer 1
    k_scatter<HID><<<(N_EDGES * (HID / 4) + 255) / 256, 256, 0, stream>>>(src, dst, h0, agg1);
    k_sage<HID, 4><<<N_NODES / 4, HID, 0, stream>>>(h0, agg1, deg, Wself1, Wneigh1, b1, h1, 0);
    // decoder projection + f16 packing + side outputs
    k_linear<HID, 4><<<N_NODES / 4, HID, 0, stream>>>(h1, Wdec, bdec, hw);
    k_pack<<<(N_NODES * HID + 255) / 256, 256, 0, stream>>>(h1, hw, h16, hw16, out_h);
    k_labels<<<(N_NODES + 255) / 256, 256, 0, stream>>>(labels, out_lab);
    // adj = softmax(relu(hW @ h^T))
    k_score_wmma<<<N_NODES / 16, 256, 0, stream>>>(hw16, h16, out_adj, rowmax);
    k_softmax<<<N_NODES, 256, 0, stream>>>(out_adj, rowmax);
}
